// GraphTransformerNet_67164698575102
// MI455X (gfx1250) — compile-verified, matrix operand order checked by reading, not run on hardware
//
#include <hip/hip_runtime.h>
#include <hip/hip_bf16.h>
#include <stdint.h>

// ---------------------------------------------------------------------------
// GraphTransformer for MI455X (gfx1250, wave32).
// Dense GEMMs via v_wmma_f32_16x16x32_bf16 with B-panel staged in LDS
// (async global->LDS copy, ASYNCcnt-tracked); edge softmax-aggregation on
// bf16 q/k/v (L2-resident) with sorted-dst register accumulation + f32
// global atomics.
// ---------------------------------------------------------------------------

typedef __attribute__((ext_vector_type(16))) __bf16 v16bf;
typedef __attribute__((ext_vector_type(8)))  float  v8f;
typedef __attribute__((ext_vector_type(8)))  unsigned int v8u;

union FragBF { v8u u; v16bf b; uint4 q[2]; };

#if __has_builtin(__builtin_amdgcn_global_load_async_to_lds_b128)
#define ASYNC_LDS_COPY 1
// exact parameter type per hipcc diagnostic: vector_size(16) int, AS1 / AS3
typedef int v4i_vs __attribute__((vector_size(16)));
typedef __attribute__((address_space(1))) v4i_vs* as1_v4i;
typedef __attribute__((address_space(3))) v4i_vs* as3_v4i;
#endif

__device__ __forceinline__ uint16_t f2bf(float f) {
  uint32_t u = __float_as_uint(f);
  u += 0x7FFFu + ((u >> 16) & 1u);      // round-to-nearest-even
  return (uint16_t)(u >> 16);
}
__device__ __forceinline__ float bf2f(uint32_t bits16) {
  return __uint_as_float(bits16 << 16);
}

// ---------------------------------------------------------------------------
// Elementwise helpers
// ---------------------------------------------------------------------------
__global__ __launch_bounds__(256) void zero_f32_kernel(float* __restrict__ p, long n) {
  long i = (long)blockIdx.x * blockDim.x + threadIdx.x;
  if (i < n) p[i] = 0.0f;
}

// Wt[n*K + k] = bf16(W[k*N + n])  (transpose so B-fragments load like A-fragments)
__global__ __launch_bounds__(256)
void cvt_transpose_kernel(const float* __restrict__ W, uint16_t* __restrict__ Wt,
                          int K, int Ncol) {
  long i = (long)blockIdx.x * blockDim.x + threadIdx.x;
  if (i >= (long)K * Ncol) return;
  int kk = (int)(i / Ncol), n = (int)(i % Ncol);
  Wt[(long)n * K + kk] = f2bf(W[i]);
}

// h = emb[feat]; also bf16 copy for the first WMMA GEMM
__global__ __launch_bounds__(256)
void embed_kernel(const int* __restrict__ feat, const float* __restrict__ emb,
                  float* __restrict__ h, uint16_t* __restrict__ hbf, long N) {
  long i = (long)blockIdx.x * blockDim.x + threadIdx.x;
  if (i >= N * 128) return;
  int node = (int)(i >> 7), ch = (int)(i & 127);
  float v = emb[(long)feat[node] * 128 + ch];
  h[i] = v;
  hbf[i] = f2bf(v);
}

// attn_bf = bf16( wV / (z[head] + 1e-6) )
__global__ __launch_bounds__(256)
void attn_div_kernel(const float* __restrict__ wV, const float* __restrict__ z,
                     uint16_t* __restrict__ attnbf, long N) {
  long i = (long)blockIdx.x * blockDim.x + threadIdx.x;
  if (i >= N * 128) return;
  int node = (int)(i >> 7), ch = (int)(i & 127), head = ch >> 4;
  float zz = z[(long)node * 8 + head] + 1e-6f;
  attnbf[i] = f2bf(wV[i] / zz);
}

// ---------------------------------------------------------------------------
// WMMA GEMM: C[M x Ncols] = A_bf16[M x K] * Bt_bf16[Ncols x K]^T  (+bias/relu)
// block = 256 thr = 8 waves; block tile 128 rows x 128 cols; wave: 16 rows.
// B panel staged per 128-K slice in LDS (pad stride 136 -> conflict-free
// ds_load_b128: lane L hits banks (4L+off)%64). K must be a multiple of 128.
// mode bits: 1=add bias, 2=relu, 4=emit bf16 (else f32)
// ---------------------------------------------------------------------------
#define BK 128
#define BS_STRIDE 136   // halves: 128 + 8 pad

__global__ __launch_bounds__(256)
void gemm_bf16_kernel(const uint16_t* __restrict__ A, const uint16_t* __restrict__ Bt,
                      const float* __restrict__ bias, void* __restrict__ C,
                      int M, int K, int Ncols, int mode) {
  __shared__ uint16_t Bs[128 * BS_STRIDE];

  const int lane = threadIdx.x & 31;
  const int wave = threadIdx.x >> 5;
  const int tileRow = blockIdx.x * 128 + wave * 16;
  const int colBase = blockIdx.y * 128;
  const bool active = tileRow < M;             // wave-uniform

  v8f acc[8];
#pragma unroll
  for (int t = 0; t < 8; ++t) acc[t] = (v8f){0,0,0,0,0,0,0,0};

  const int mrow = lane & 15;                  // M within tile
  const int kg   = (lane >> 4) * 8;            // lanes 16-31 hold K+8 per ISA layout
  const uint16_t* aRow = A + (long)(tileRow + mrow) * K;

  // slab-copy assignment: thread -> (col, 64-half K chunk)
  const int lcol  = threadIdx.x >> 1;          // 0..127
  const int lhalf = (threadIdx.x & 1) * 64;    // 0 or 64

  for (int ks = 0; ks < K; ks += BK) {
    // cooperative stage of B slab [128 cols x 128 K] into LDS
    {
      const uint16_t* gsrc = Bt + (long)(colBase + lcol) * K + ks + lhalf;
      uint16_t* ldst = Bs + lcol * BS_STRIDE + lhalf;
#ifdef ASYNC_LDS_COPY
#pragma unroll
      for (int i = 0; i < 8; ++i)
        __builtin_amdgcn_global_load_async_to_lds_b128(
            (as1_v4i)(void*)(gsrc + i * 8), (as3_v4i)(void*)(ldst + i * 8), 0, 0);
#if __has_builtin(__builtin_amdgcn_s_wait_asynccnt)
      __builtin_amdgcn_s_wait_asynccnt(0);
#else
      asm volatile("s_wait_asynccnt 0x0" ::: "memory");
#endif
#else
      uint4 tmp[8];                            // distinct temps: one clause, one wait
#pragma unroll
      for (int i = 0; i < 8; ++i) tmp[i] = *(const uint4*)(gsrc + i * 8);
#pragma unroll
      for (int i = 0; i < 8; ++i) *(uint4*)(ldst + i * 8) = tmp[i];
#endif
    }
    __syncthreads();

    if (active) {
      for (int kk = 0; kk < BK; kk += 32) {
        FragBF af;
        const uint16_t* ap = aRow + ks + kk + kg;
        __builtin_prefetch(ap + 32, 0, 1);     // global_prefetch next K-chunk
        af.q[0] = *(const uint4*)(ap);         // K = kg+0..7   (pairs contiguous)
        af.q[1] = *(const uint4*)(ap + 16);    // K = kg+16..23

        FragBF bfr[8];                         // all 8 tiles' B frags from LDS first
#pragma unroll
        for (int t = 0; t < 8; ++t) {
          const uint16_t* bp = Bs + (t * 16 + mrow) * BS_STRIDE + kk + kg;
          bfr[t].q[0] = *(const uint4*)(bp);
          bfr[t].q[1] = *(const uint4*)(bp + 16);
        }
#pragma unroll
        for (int t = 0; t < 8; ++t)
          acc[t] = __builtin_amdgcn_wmma_f32_16x16x32_bf16(
              false, af.b, false, bfr[t].b, (short)0, acc[t], false, false);
      }
    }
    __syncthreads();
  }

  if (!active) return;
  // C layout: VGPR r -> lanes 0-15: (M=r, N=lane); lanes 16-31: (M=r+8, N=lane-16)
  const int nOff = lane & 15;
  const int m0   = (lane >> 4) * 8;
#pragma unroll
  for (int t = 0; t < 8; ++t) {
    const int col = colBase + t * 16 + nOff;
    const float bv = (mode & 1) ? bias[col] : 0.0f;
#pragma unroll
    for (int r = 0; r < 8; ++r) {
      float v = acc[t][r] + bv;
      if (mode & 2) v = fmaxf(v, 0.0f);
      const long idx = (long)(tileRow + m0 + r) * Ncols + col;
      if (mode & 4) ((uint16_t*)C)[idx] = f2bf(v);
      else          ((float*)C)[idx]    = v;
    }
  }
}

// ---------------------------------------------------------------------------
// Edge attention on bf16 q/k/v (q+k+v = 76.8MB -> L2-resident at 192MB L2).
// One wave per 64 sorted-by-dst edges; register accumulation, flush with
// global f32 atomics on dst change. Lane owns 4 channels; head = lane>>2.
// ---------------------------------------------------------------------------
__global__ __launch_bounds__(256)
void edge_attn_kernel(const uint16_t* __restrict__ q, const uint16_t* __restrict__ k,
                      const uint16_t* __restrict__ v, const int* __restrict__ esrc,
                      const int* __restrict__ edst, float* __restrict__ wV,
                      float* __restrict__ z, long E) {
  const int lane = threadIdx.x & 31;
  const long wid = (long)blockIdx.x * 8 + (threadIdx.x >> 5);
  long e0 = wid * 64;
  if (e0 >= E) return;
  long e1 = e0 + 64; if (e1 > E) e1 = E;

  const int c = lane << 2;
  const int head = lane >> 2;
  float a0 = 0.f, a1 = 0.f, a2 = 0.f, a3 = 0.f, az = 0.f;
  int cur = edst[e0];

  for (long e = e0; e < e1; ++e) {
    const int s = esrc[e];
    const int d = edst[e];
    if (d != cur) {
      unsafeAtomicAdd(&wV[(long)cur * 128 + c + 0], a0);
      unsafeAtomicAdd(&wV[(long)cur * 128 + c + 1], a1);
      unsafeAtomicAdd(&wV[(long)cur * 128 + c + 2], a2);
      unsafeAtomicAdd(&wV[(long)cur * 128 + c + 3], a3);
      if ((lane & 3) == 0) unsafeAtomicAdd(&z[(long)cur * 8 + head], az);
      a0 = a1 = a2 = a3 = az = 0.f;
      cur = d;
    }
    const uint2 kp = *(const uint2*)(k + (long)s * 128 + c);
    const uint2 qp = *(const uint2*)(q + (long)d * 128 + c);
    float p = bf2f(kp.x & 0xffffu) * bf2f(qp.x & 0xffffu)
            + bf2f(kp.x >> 16)     * bf2f(qp.x >> 16)
            + bf2f(kp.y & 0xffffu) * bf2f(qp.y & 0xffffu)
            + bf2f(kp.y >> 16)     * bf2f(qp.y >> 16);
    p += __shfl_xor(p, 1, 32);
    p += __shfl_xor(p, 2, 32);
    const float sc = __expf(fminf(fmaxf(p * 0.25f, -5.0f), 5.0f)); // 1/sqrt(16)=0.25
    const uint2 vp = *(const uint2*)(v + (long)s * 128 + c);
    a0 += sc * bf2f(vp.x & 0xffffu);
    a1 += sc * bf2f(vp.x >> 16);
    a2 += sc * bf2f(vp.y & 0xffffu);
    a3 += sc * bf2f(vp.y >> 16);
    az += sc;
  }
  unsafeAtomicAdd(&wV[(long)cur * 128 + c + 0], a0);
  unsafeAtomicAdd(&wV[(long)cur * 128 + c + 1], a1);
  unsafeAtomicAdd(&wV[(long)cur * 128 + c + 2], a2);
  unsafeAtomicAdd(&wV[(long)cur * 128 + c + 3], a3);
  if ((lane & 3) == 0) unsafeAtomicAdd(&z[(long)cur * 8 + head], az);
}

// ---------------------------------------------------------------------------
// residual + LayerNorm over 128 channels; one wave per node (4 ch / lane).
// Writes f32 h (in place) and a bf16 copy for the next WMMA GEMM.
// ---------------------------------------------------------------------------
__global__ __launch_bounds__(256)
void residual_ln_kernel(float* __restrict__ h, const float* __restrict__ y,
                        const float* __restrict__ g, const float* __restrict__ b,
                        uint16_t* __restrict__ hbf, long N) {
  const int lane = threadIdx.x & 31;
  const long node = (long)blockIdx.x * 8 + (threadIdx.x >> 5);
  if (node >= N) return;
  const long base = node * 128 + lane * 4;
  const float4 hv = *(const float4*)(h + base);
  const float4 yv = *(const float4*)(y + base);
  float t0 = hv.x + yv.x, t1 = hv.y + yv.y, t2 = hv.z + yv.z, t3 = hv.w + yv.w;

  float s = t0 + t1 + t2 + t3;
#pragma unroll
  for (int o = 1; o < 32; o <<= 1) s += __shfl_xor(s, o, 32);
  const float mu = s * (1.0f / 128.0f);

  const float d0 = t0 - mu, d1 = t1 - mu, d2 = t2 - mu, d3 = t3 - mu;
  float sq = d0 * d0 + d1 * d1 + d2 * d2 + d3 * d3;
#pragma unroll
  for (int o = 1; o < 32; o <<= 1) sq += __shfl_xor(sq, o, 32);
  const float inv = rsqrtf(sq * (1.0f / 128.0f) + 1e-5f);

  const int c = lane * 4;
  const float o0 = d0 * inv * g[c + 0] + b[c + 0];
  const float o1 = d1 * inv * g[c + 1] + b[c + 1];
  const float o2 = d2 * inv * g[c + 2] + b[c + 2];
  const float o3 = d3 * inv * g[c + 3] + b[c + 3];
  *(float4*)(h + base) = make_float4(o0, o1, o2, o3);
  hbf[base + 0] = f2bf(o0); hbf[base + 1] = f2bf(o1);
  hbf[base + 2] = f2bf(o2); hbf[base + 3] = f2bf(o3);
}

// ---------------------------------------------------------------------------
// MLP readout 128->64->32->16; one wave per node, shuffle-chained layers.
// ---------------------------------------------------------------------------
__global__ __launch_bounds__(256)
void readout_kernel(const float* __restrict__ h,
                    const float* __restrict__ mW0, const float* __restrict__ mb0,
                    const float* __restrict__ mW1, const float* __restrict__ mb1,
                    const float* __restrict__ mW2, const float* __restrict__ mb2,
                    float* __restrict__ out, long N) {
  const int lane = threadIdx.x & 31;
  const long node = (long)blockIdx.x * 8 + (threadIdx.x >> 5);
  if (node >= N) return;
  const float* hr = h + node * 128;

  float r1[2];                                      // lane owns outputs lane, lane+32
#pragma unroll
  for (int sblk = 0; sblk < 2; ++sblk) {
    const int o = lane + 32 * sblk;
    float a = mb0[o];
    for (int kk = 0; kk < 128; ++kk) a += hr[kk] * mW0[kk * 64 + o];
    r1[sblk] = fmaxf(a, 0.0f);
  }

  float a1 = mb1[lane];                             // 32 outputs, one per lane
  for (int j = 0; j < 64; ++j)
    a1 += __shfl(r1[j >> 5], j & 31, 32) * mW1[j * 32 + lane];
  a1 = fmaxf(a1, 0.0f);

  const int oc = lane & 15;                         // 16 outputs
  float a2 = mb2[oc];
  for (int j = 0; j < 32; ++j)
    a2 += __shfl(a1, j, 32) * mW2[j * 16 + oc];
  if (lane < 16) out[node * 16 + lane] = a2;
}

// ---------------------------------------------------------------------------
// Host orchestration
// ---------------------------------------------------------------------------
extern "C" void kernel_launch(void* const* d_in, const int* in_sizes, int n_in,
                              void* d_out, int out_size, void* d_ws, size_t ws_size,
                              hipStream_t stream) {
  const long N = in_sizes[0];
  const long E = in_sizes[1];
  const int LAYERS = 4;

  const int*   feat  = (const int*)d_in[0];
  const int*   esrc  = (const int*)d_in[1];
  const int*   edst  = (const int*)d_in[2];
  const float* emb   = (const float*)d_in[3];
  const float* Wq    = (const float*)d_in[4];
  const float* Wk    = (const float*)d_in[5];
  const float* Wv    = (const float*)d_in[6];
  const float* Wo    = (const float*)d_in[7];
  const float* bo    = (const float*)d_in[8];
  const float* ln1g  = (const float*)d_in[9];
  const float* ln1b  = (const float*)d_in[10];
  const float* W1    = (const float*)d_in[11];
  const float* b1    = (const float*)d_in[12];
  const float* W2    = (const float*)d_in[13];
  const float* b2    = (const float*)d_in[14];
  const float* ln2g  = (const float*)d_in[15];
  const float* ln2b  = (const float*)d_in[16];
  const float* mW0   = (const float*)d_in[17];
  const float* mb0   = (const float*)d_in[18];
  const float* mW1   = (const float*)d_in[19];
  const float* mb1   = (const float*)d_in[20];
  const float* mW2   = (const float*)d_in[21];
  const float* mb2   = (const float*)d_in[22];
  float* outp = (float*)d_out;

  // ---- workspace layout (overlaid where lifetimes allow) ----
  char* ws = (char*)d_ws;
  size_t off = 0;
  auto alloc = [&](size_t bytes) { size_t o = off; off = (off + bytes + 255) & ~(size_t)255; return o; };
  float*    h     = (float*)   (ws + alloc(N * 128 * 4));
  uint16_t* hbf   = (uint16_t*)(ws + alloc(N * 128 * 2));
  uint16_t* qb    = (uint16_t*)(ws + alloc(N * 256 * 2));   // q bf16 / ffn-hidden bf16 [N x 256]
  uint16_t* kb    = (uint16_t*)(ws + alloc(N * 128 * 4));   // k bf16 / tmp128 f32
  uint16_t* vb    = (uint16_t*)(ws + alloc(N * 128 * 4));   // v bf16 / attn bf16
  float*    wV    = (float*)   (ws + alloc(N * 128 * 4));
  float*    zb    = (float*)   (ws + alloc(N * 8 * 4));
  uint16_t* wbf   = (uint16_t*)(ws + alloc((size_t)LAYERS * 131072 * 2));
  uint16_t* ffnbf = qb;                 // [N x 256] bf16, overlays q
  float*    t128  = (float*)kb;         // [N x 128] f32,  overlays k
  uint16_t* atbf  = vb;                 // [N x 128] bf16, overlays v

  // ---- weights -> bf16, transposed to [n][k] (deterministic, re-done each call) ----
  const int CVT128 = (128 * 128 + 255) / 256, CVT256 = (128 * 256 + 255) / 256;
  for (int l = 0; l < LAYERS; ++l) {
    uint16_t* wl = wbf + (size_t)l * 131072;
    cvt_transpose_kernel<<<CVT128, 256, 0, stream>>>(Wq + (long)l*128*128, wl +     0, 128, 128);
    cvt_transpose_kernel<<<CVT128, 256, 0, stream>>>(Wk + (long)l*128*128, wl + 16384, 128, 128);
    cvt_transpose_kernel<<<CVT128, 256, 0, stream>>>(Wv + (long)l*128*128, wl + 32768, 128, 128);
    cvt_transpose_kernel<<<CVT128, 256, 0, stream>>>(Wo + (long)l*128*128, wl + 49152, 128, 128);
    cvt_transpose_kernel<<<CVT256, 256, 0, stream>>>(W1 + (long)l*128*256, wl + 65536, 128, 256);
    cvt_transpose_kernel<<<CVT256, 256, 0, stream>>>(W2 + (long)l*256*128, wl + 98304, 256, 128);
  }

  const long NEL = N * 128;
  const int EW_BLK   = (int)((NEL + 255) / 256);
  const int NODE_BLK = (int)((N + 7) / 8);
  const dim3 G128((unsigned)((N + 127) / 128), 1);
  const dim3 G256((unsigned)((N + 127) / 128), 2);
  const int EDGE_BLK = (int)(((E + 63) / 64 + 7) / 8);

  embed_kernel<<<EW_BLK, 256, 0, stream>>>(feat, emb, h, hbf, N);

  for (int l = 0; l < LAYERS; ++l) {
    const uint16_t* wl = wbf + (size_t)l * 131072;
    // QKV projections (no bias, bf16 out -> L2-resident edge working set)
    gemm_bf16_kernel<<<G128, 256, 0, stream>>>(hbf, wl +     0, nullptr, qb, (int)N, 128, 128, 4);
    gemm_bf16_kernel<<<G128, 256, 0, stream>>>(hbf, wl + 16384, nullptr, kb, (int)N, 128, 128, 4);
    gemm_bf16_kernel<<<G128, 256, 0, stream>>>(hbf, wl + 32768, nullptr, vb, (int)N, 128, 128, 4);
    // Edge softmax-aggregation
    zero_f32_kernel<<<EW_BLK, 256, 0, stream>>>(wV, NEL);
    zero_f32_kernel<<<(int)((N * 8 + 255) / 256), 256, 0, stream>>>(zb, N * 8);
    edge_attn_kernel<<<EDGE_BLK, 256, 0, stream>>>(qb, kb, vb, esrc, edst, wV, zb, E);
    attn_div_kernel<<<EW_BLK, 256, 0, stream>>>(wV, zb, atbf, N);
    // O-proj + bias -> tmp128; residual + LN1 -> h/hbf
    gemm_bf16_kernel<<<G128, 256, 0, stream>>>(atbf, wl + 49152, bo + (long)l*128, t128, (int)N, 128, 128, 1);
    residual_ln_kernel<<<NODE_BLK, 256, 0, stream>>>(h, t128, ln1g + (long)l*128, ln1b + (long)l*128, hbf, N);
    // FFN: (bias+relu, bf16 out) then (bias, f32 out); residual + LN2
    gemm_bf16_kernel<<<G256, 256, 0, stream>>>(hbf, wl + 65536, b1 + (long)l*256, ffnbf, (int)N, 128, 256, 1|2|4);
    gemm_bf16_kernel<<<G128, 256, 0, stream>>>(ffnbf, wl + 98304, b2 + (long)l*128, t128, (int)N, 256, 128, 1);
    residual_ln_kernel<<<NODE_BLK, 256, 0, stream>>>(h, t128, ln2g + (long)l*128, ln2b + (long)l*128, hbf, N);
  }

  readout_kernel<<<NODE_BLK, 256, 0, stream>>>(h, mW0, mb0, mW1, mb1, mW2, mb2, outp, N);
}
